// YOWO_74749610819734
// MI455X (gfx1250) — compile-verified, matrix operand order checked by reading, not run on hardware
//
#include <hip/hip_runtime.h>

// ---------------- problem constants (match reference) ----------------
#define BATCH       32
#define M_ANCH      20480          // (512/8)^2 * 5
#define NCLS        80
#define TOPK        1000
#define CAP         4096           // candidate capacity per batch
#define CAND_THRESH 0.02f          // gather slightly below CONF_THRESH
#define CONF_THRESH 0.05f
#define NMS_THRESH  0.6f

// score-kernel tiling
#define WAVES            4         // 128-thread blocks
#define TILE_ANCH        (WAVES * 16)   // 64 anchors per tile
#define TILES_PER_BLOCK  4
#define RS               84        // padded LDS row stride in floats (bank-conflict free, 16B aligned)
#define NSTAGE           10        // async b128 issues per wave per tile (16 rows * 20 float4 / 32 lanes)

typedef __attribute__((ext_vector_type(16))) _Float16 v16h;
typedef __attribute__((ext_vector_type(8)))  float    v8f;

// ---- CDNA5 async-to-LDS path (probe-guarded; fallback = direct loads) ----
#if __has_builtin(__builtin_amdgcn_global_load_async_to_lds_b128)
#define YOWO_ASYNC 1
typedef int v4i_ __attribute__((ext_vector_type(4)));
typedef __attribute__((address_space(1))) v4i_* as1_v4i;
typedef __attribute__((address_space(3))) v4i_* as3_v4i;
#if __has_builtin(__builtin_amdgcn_s_wait_asynccnt)
#define WAIT_ASYNC(n) __builtin_amdgcn_s_wait_asynccnt(n)
#else
#define WAIT_ASYNC(n) asm volatile("s_wait_asynccnt %0" :: "i"(n) : "memory")
#endif
#endif

__device__ __forceinline__ float sigmoidf_(float x) { return 1.0f / (1.0f + __expf(-x)); }

// ---------------- kernel 0: zero per-batch candidate counters ----------------
__global__ void yowo_init_kernel(unsigned int* __restrict__ cnt) {
    if (threadIdx.x < BATCH) cnt[threadIdx.x] = 0u;
}

// ---------------- kernel 1: fused sigmoid*softmax + candidate compaction ----
// Persistent double-buffered pipeline: async-stage tile t+1 into LDS while the
// matrix unit + VALU chew on tile t.  One wave owns 16 anchors per tile; the
// softmax row-sum runs on 3x v_wmma_f32_16x16x32_f16 with B = ones (row-sum is
// K-permutation invariant, so lanes stage/load contiguous 16-float spans).
__global__ __launch_bounds__(128)
void yowo_score_kernel(const float* __restrict__ conf,
                       const float* __restrict__ cls,
                       unsigned int* __restrict__ cnt,
                       unsigned long long* __restrict__ cand)
{
    const int tid   = threadIdx.x;
    const int wave  = tid >> 5;
    const int lane  = tid & 31;
    const int r     = lane & 15;    // anchor row within the wave's 16-row tile
    const int h     = lane >> 4;    // which half of each 32-wide K chunk
    const int b     = blockIdx.y;
    const int tile0 = blockIdx.x * TILES_PER_BLOCK;

#if defined(YOWO_ASYNC)
    __shared__ float sbuf[2][WAVES][16 * RS];     // 2 x 4 x 5376 B = 43 KB
    // fixed per-lane staging pattern: float4 slot f covers row f/20, quad f%20
    int srow[NSTAGE], sk4[NSTAGE];
#pragma unroll
    for (int i = 0; i < NSTAGE; ++i) {
        int f = i * 32 + lane;
        srow[i] = f / 20;
        sk4[i]  = f % 20;
    }
    auto stage = [&](int buf, int t) {
        const float* gw = cls + ((size_t)b * M_ANCH
                                 + (size_t)(tile0 + t) * TILE_ANCH + wave * 16) * NCLS;
#pragma unroll
        for (int i = 0; i < NSTAGE; ++i) {
            const float* gp = gw + srow[i] * NCLS + sk4[i] * 4;
            float*       lp = &sbuf[buf][wave][srow[i] * RS + sk4[i] * 4];
            __builtin_amdgcn_global_load_async_to_lds_b128(
                (as1_v4i)gp, (as3_v4i)lp, 0, 0);
        }
    };
    stage(0, 0);
#endif

    __shared__ float ssum[WAVES][16];

    for (int t = 0; t < TILES_PER_BLOCK; ++t) {
        const int m = (tile0 + t) * TILE_ANCH + wave * 16 + r;   // anchor index
        __builtin_prefetch(conf + (size_t)b * M_ANCH + m, 0, 0); // global_prefetch_b8

#if defined(YOWO_ASYNC)
        if (t + 1 < TILES_PER_BLOCK) {
            stage((t + 1) & 1, t + 1);
            WAIT_ASYNC(NSTAGE);     // tile t fully landed; t+1 may be in flight
        } else {
            WAIT_ASYNC(0);
        }
#endif

        // classes covered by this lane: h==0 -> [0,16)+[32,48)+[64,80)
        //                               h==1 -> [16,32)+[48,64)+pad(zeros)
        float lg[3][16];
#if defined(YOWO_ASYNC)
        const float* wrow = &sbuf[t & 1][wave][r * RS];
#pragma unroll
        for (int c = 0; c < 3; ++c) {
            if (c == 2 && h == 1) {
#pragma unroll
                for (int j = 0; j < 16; ++j) lg[c][j] = -1e30f;   // zero-pad (exp -> 0)
            } else {
                const float4* p = (const float4*)(wrow + c * 32 + h * 16);
#pragma unroll
                for (int q = 0; q < 4; ++q) {
                    float4 v = p[q];
                    lg[c][q * 4 + 0] = v.x; lg[c][q * 4 + 1] = v.y;
                    lg[c][q * 4 + 2] = v.z; lg[c][q * 4 + 3] = v.w;
                }
            }
        }
#else
        const float* rowp = cls + ((size_t)b * M_ANCH + m) * NCLS;
#pragma unroll
        for (int c = 0; c < 3; ++c) {
            if (c == 2 && h == 1) {
#pragma unroll
                for (int j = 0; j < 16; ++j) lg[c][j] = -1e30f;
            } else {
                const float4* p = (const float4*)(rowp + c * 32 + h * 16);
#pragma unroll
                for (int q = 0; q < 4; ++q) {
                    float4 v = p[q];
                    lg[c][q * 4 + 0] = v.x; lg[c][q * 4 + 1] = v.y;
                    lg[c][q * 4 + 2] = v.z; lg[c][q * 4 + 3] = v.w;
                }
            }
        }
#endif

        // row max across the lane pair (wave32: xor 16)
        float rm = -1e30f;
#pragma unroll
        for (int c = 0; c < 3; ++c)
#pragma unroll
            for (int j = 0; j < 16; ++j) rm = fmaxf(rm, lg[c][j]);
        rm = fmaxf(rm, __shfl_xor(rm, 16, 32));

        // exp -> f16 A tiles; row sums via 3 chained WMMAs with B = ones
        v16h a[3];
#pragma unroll
        for (int c = 0; c < 3; ++c)
#pragma unroll
            for (int j = 0; j < 16; ++j) a[c][j] = (_Float16)__expf(lg[c][j] - rm);

        v16h ones;
#pragma unroll
        for (int j = 0; j < 16; ++j) ones[j] = (_Float16)1.0f;

        v8f acc = {};
#pragma unroll
        for (int c = 0; c < 3; ++c)
            acc = __builtin_amdgcn_wmma_f32_16x16x32_f16(
                      false, a[c], false, ones, (short)0, acc, false, false);

        // D layout: lanes 0-15 hold rows 0-7 in acc[0..7]; lanes 16-31 rows 8-15.
        if (lane == 0) {
#pragma unroll
            for (int k = 0; k < 8; ++k) ssum[wave][k] = acc[k];
        }
        if (lane == 16) {
#pragma unroll
            for (int k = 0; k < 8; ++k) ssum[wave][8 + k] = acc[k];
        }
        __syncthreads();
        const float inv_sum = 1.0f / ssum[wave][r];   // sum >= 1 (contains exp(0))
        const float scale   = sigmoidf_(conf[(size_t)b * M_ANCH + m]) * inv_sum;

        // threshold-compact candidates: key = (f32 score bits << 32) | flat index
#pragma unroll
        for (int c = 0; c < 3; ++c)
#pragma unroll
            for (int j = 0; j < 16; ++j) {
                float s = scale * (float)a[c][j];     // padded entries are exactly 0
                if (s > CAND_THRESH) {
                    unsigned int k    = (unsigned)(c * 32 + h * 16 + j);
                    unsigned int flat = (unsigned)m * NCLS + k;
                    unsigned int pos  = atomicAdd(&cnt[b], 1u);
                    if (pos < CAP)
                        cand[(size_t)b * CAP + pos] =
                            ((unsigned long long)__float_as_uint(s) << 32) | flat;
                }
            }
        __syncthreads();
    }
}

// ---------------- kernel 2: per-batch bitonic top-1000 + box decode ---------
__global__ __launch_bounds__(1024)
void yowo_topk_decode_kernel(const float* __restrict__ reg,
                             const unsigned int* __restrict__ cnt,
                             const unsigned long long* __restrict__ cand,
                             float* __restrict__ wboxes,
                             float* __restrict__ wscores,
                             int*   __restrict__ wlabels,
                             int*   __restrict__ wkeep0,
                             float* __restrict__ out)
{
    __shared__ unsigned long long keys[CAP];
    const int b = blockIdx.x, tid = threadIdx.x;

    int n = (int)cnt[b]; if (n > CAP) n = CAP;
    for (int i = tid; i < CAP; i += 1024)
        keys[i] = (i < n) ? cand[(size_t)b * CAP + i] : 0ull;
    __syncthreads();

    // bitonic sort, descending (keys are distinct: idx in low bits)
    for (int k = 2; k <= CAP; k <<= 1)
        for (int j = k >> 1; j > 0; j >>= 1) {
            for (int i = tid; i < CAP; i += 1024) {
                int ixj = i ^ j;
                if (ixj > i) {
                    unsigned long long x = keys[i], y = keys[ixj];
                    bool up = ((i & k) == 0);
                    if (up ? (x < y) : (x > y)) { keys[i] = y; keys[ixj] = x; }
                }
            }
            __syncthreads();
        }

    if (tid < TOPK) {
        unsigned long long key = keys[tid];
        float score        = __uint_as_float((unsigned)(key >> 32));
        unsigned int flat  = (unsigned)key;
        if (key == 0ull) { score = 0.0f; flat = 0u; }
        int anchor = (int)(flat / NCLS);
        int label  = (int)(flat % NCLS);

        // regenerate anchor geometry
        int slot = anchor % 5, cell = anchor / 5;
        float ax = (float)((cell & 63) * 8);
        float ay = (float)((cell >> 6) * 8);
        const float AW[5] = {16.f, 32.f, 64.f, 128.f, 256.f};
        const float AH[5] = {16.f, 64.f, 32.f, 128.f, 256.f};
        float aw = AW[slot], ah = AH[slot];

        const float* rp = reg + ((size_t)b * M_ANCH + anchor) * 4;
        float sx = sigmoidf_(rp[0]) * 8.0f + ax;
        float sy = sigmoidf_(rp[1]) * 8.0f + ay;
        float bw = __expf(rp[2]) * aw;
        float bh = __expf(rp[3]) * ah;
        const float inv_img = 1.0f / 512.0f;
        float x1 = fminf(fmaxf((sx - 0.5f * bw) * inv_img, 0.f), 1.f);
        float y1 = fminf(fmaxf((sy - 0.5f * bh) * inv_img, 0.f), 1.f);
        float x2 = fminf(fmaxf((sx + 0.5f * bw) * inv_img, 0.f), 1.f);
        float y2 = fminf(fmaxf((sy + 0.5f * bh) * inv_img, 0.f), 1.f);

        size_t o = (size_t)b * TOPK + tid;
        wboxes[o * 4 + 0] = x1; wboxes[o * 4 + 1] = y1;
        wboxes[o * 4 + 2] = x2; wboxes[o * 4 + 3] = y2;
        wscores[o] = score;
        wlabels[o] = label;
        wkeep0[o]  = (score > CONF_THRESH) ? 1 : 0;
        out[(size_t)BATCH * TOPK * 5 + o] = (float)label;   // labels section
    }
}

// ---------------- kernel 3: per-batch greedy NMS + final outputs ------------
__global__ __launch_bounds__(1024)
void yowo_nms_kernel(const float* __restrict__ wboxes,
                     const float* __restrict__ wscores,
                     const int*   __restrict__ wlabels,
                     const int*   __restrict__ wkeep0,
                     float* __restrict__ out)
{
    __shared__ float x1s[TOPK], y1s[TOPK], x2s[TOPK], y2s[TOPK], ars[TOPK], scs[TOPK];
    __shared__ int   labs[TOPK], sup[TOPK], kp[TOPK];
    const int b = blockIdx.x, tid = threadIdx.x;

    for (int i = tid; i < TOPK; i += 1024) {
        size_t o = (size_t)b * TOPK + i;
        float x1 = wboxes[o * 4 + 0], y1 = wboxes[o * 4 + 1];
        float x2 = wboxes[o * 4 + 2], y2 = wboxes[o * 4 + 3];
        x1s[i] = x1; y1s[i] = y1; x2s[i] = x2; y2s[i] = y2;
        ars[i] = (x2 - x1) * (y2 - y1);
        scs[i] = wscores[o]; labs[i] = wlabels[o];
        kp[i] = wkeep0[o]; sup[i] = 0;
    }
    __syncthreads();

    for (int i = 0; i < TOPK - 1; ++i) {
        bool act = (kp[i] != 0) && (sup[i] == 0);
        if (act) {
            float xi1 = x1s[i], yi1 = y1s[i], xi2 = x2s[i], yi2 = y2s[i], ai = ars[i];
            int li = labs[i];
            for (int j = i + 1 + tid; j < TOPK; j += 1024) {
                if (labs[j] == li && sup[j] == 0) {
                    float xx1 = fmaxf(xi1, x1s[j]), yy1 = fmaxf(yi1, y1s[j]);
                    float xx2 = fminf(xi2, x2s[j]), yy2 = fminf(yi2, y2s[j]);
                    float w  = fmaxf(1e-28f, xx2 - xx1);
                    float hh = fmaxf(1e-28f, yy2 - yy1);
                    float inter = w * hh;
                    float iou = inter / (ai + ars[j] - inter + 1e-14f);
                    if (iou > NMS_THRESH) sup[j] = 1;
                }
            }
        }
        __syncthreads();
    }

    for (int i = tid; i < TOPK; i += 1024) {
        float kv = (kp[i] != 0 && sup[i] == 0) ? 1.0f : 0.0f;
        size_t o = (size_t)b * TOPK + i;
        out[o * 5 + 0] = x1s[i] * kv;
        out[o * 5 + 1] = y1s[i] * kv;
        out[o * 5 + 2] = x2s[i] * kv;
        out[o * 5 + 3] = y2s[i] * kv;
        out[o * 5 + 4] = scs[i] * kv;
        out[(size_t)BATCH * TOPK * 6 + o] = kv;             // keep section
    }
}

// ---------------- host launcher ----------------
extern "C" void kernel_launch(void* const* d_in, const int* in_sizes, int n_in,
                              void* d_out, int out_size, void* d_ws, size_t ws_size,
                              hipStream_t stream)
{
    (void)in_sizes; (void)n_in; (void)out_size; (void)ws_size;
    const float* conf = (const float*)d_in[0];
    const float* cls  = (const float*)d_in[1];
    const float* reg  = (const float*)d_in[2];
    float* out = (float*)d_out;

    char* ws = (char*)d_ws;
    unsigned int*       cnt    = (unsigned int*)ws;                        // 128 B
    unsigned long long* cand   = (unsigned long long*)(ws + 256);          // 32*4096*8 = 1 MB
    float*              wboxes = (float*)(ws + 256 + (size_t)BATCH * CAP * 8);
    float*              wscores= wboxes + (size_t)BATCH * TOPK * 4;
    int*                wlabels= (int*)(wscores + (size_t)BATCH * TOPK);
    int*                wkeep0 = wlabels + (size_t)BATCH * TOPK;

    hipLaunchKernelGGL(yowo_init_kernel, dim3(1), dim3(64), 0, stream, cnt);
    hipLaunchKernelGGL(yowo_score_kernel,
                       dim3(M_ANCH / (TILE_ANCH * TILES_PER_BLOCK), BATCH), dim3(128), 0, stream,
                       conf, cls, cnt, cand);
    hipLaunchKernelGGL(yowo_topk_decode_kernel, dim3(BATCH), dim3(1024), 0, stream,
                       reg, cnt, cand, wboxes, wscores, wlabels, wkeep0, out);
    hipLaunchKernelGGL(yowo_nms_kernel, dim3(BATCH), dim3(1024), 0, stream,
                       wboxes, wscores, wlabels, wkeep0, out);
}
